// LSTMClassifier_26886495273452
// MI455X (gfx1250) — matrix operand order repaired and checked by reading, not compiled, not run on hardware
//
#include <hip/hip_runtime.h>

typedef __attribute__((ext_vector_type(16))) _Float16 v16h;
typedef __attribute__((ext_vector_type(8)))  _Float16 v8h;
typedef __attribute__((ext_vector_type(8)))  float    v8f;
typedef __attribute__((ext_vector_type(4)))  float    v4f;

#define HID      50
#define GATES    200           // 4*HID, gate order i,f,g,o
#define NPAD     208           // 13 tiles of 16 (weight staging)
#define NT       13
#define KPAD     64            // 50 h + 3 x + 1 bias + 10 zero pad
#define GPITCH   200           // sG row pitch (floats); col n -> (n%50)*4 + n/50
#define T_STEPS  512
#define INPUT    3
#define ROWS     16            // batch rows per workgroup (WMMA M)
#define NWAVES   4
#define NTHREADS (NWAVES * 32)
#define NCLS     6
#define BIAS_K   (HID + INPUT) // 53
#define QFULL    3             // unconditional N-tiles per wave; wave0 adds nt=12

#if __has_builtin(__builtin_amdgcn_rcpf)
#define FAST_RCP(x) __builtin_amdgcn_rcpf(x)
#else
#define FAST_RCP(x) (1.f / (x))
#endif

__device__ __forceinline__ float fast_sigmoid(float x) {
    return FAST_RCP(1.f + __expf(-x));          // v_exp_f32 + v_rcp_f32
}
__device__ __forceinline__ float fast_tanh(float x) {
#if __has_builtin(__builtin_amdgcn_tanhf)
    return __builtin_amdgcn_tanhf(x);           // v_tanh_f32
#elif __has_builtin(__builtin_amdgcn_tanh_f32)
    return __builtin_amdgcn_tanh_f32(x);
#else
    return 2.f * FAST_RCP(1.f + __expf(-2.f * x)) - 1.f;
#endif
}

__global__ __launch_bounds__(NTHREADS)
void lstm_wmma_kernel(const float* __restrict__ x,
                      const float* __restrict__ W_ih,
                      const float* __restrict__ W_hh,
                      const float* __restrict__ b_ih,
                      const float* __restrict__ b_hh,
                      const float* __restrict__ W_fc,
                      const float* __restrict__ b_fc,
                      float* __restrict__ out)
{
    // Augmented weight matrix: sW[n][k] = B[k][n]; n = gate column, k in [0,KPAD)
    __shared__ __attribute__((aligned(16))) _Float16 sW[NPAD * KPAD];  // 26624 B
    __shared__ __attribute__((aligned(16))) float    sG[ROWS * GPITCH];// 12800 B
    __shared__ __attribute__((aligned(16))) _Float16 sH[ROWS * KPAD];  //  2048 B

    const int tid  = threadIdx.x;
    const int lane = tid & 31;
    // scalar wave id -> tile guards lower to s_cbranch, EXEC stays all-1s for WMMA
    const int w    = __builtin_amdgcn_readfirstlane(tid >> 5);
    const int rowBase = blockIdx.x * ROWS;

    // ---- stage augmented weight matrix into LDS (f16) ----
    for (int idx = tid; idx < NPAD * KPAD; idx += NTHREADS) {
        int n = idx >> 6, k = idx & 63;
        float v = 0.f;
        if (n < GATES) {
            if (k < HID)          v = W_hh[n * HID + k];
            else if (k < BIAS_K)  v = W_ih[n * INPUT + (k - HID)];
            else if (k == BIAS_K) v = b_ih[n] + b_hh[n];
        }
        sW[idx] = (_Float16)v;
    }
    // ---- init A rows: h=0, x staged per step, constant-1 column for bias row ----
    for (int idx = tid; idx < ROWS * KPAD; idx += NTHREADS) {
        int k = idx & 63;
        sH[idx] = (_Float16)((k == BIAS_K) ? 1.f : 0.f);
    }
    __syncthreads();

    const int m   = lane & 15;   // WMMA row (A) / column (B,C)
    const int hiK = lane >> 4;   // lane-half selector

    // ---- hoist loop-invariant B fragments into registers ----
    // wave w owns N tiles {w, w+4, w+8} (all < 12); wave 0 additionally owns nt=12
    v16h bf0[QFULL + 1] = {}, bf1[QFULL + 1] = {};
    int  sofs[QFULL + 1];        // interleaved gate-store offsets (incl. lane-half row)
    #pragma unroll
    for (int q = 0; q <= QFULL; ++q) {
        const int nt = (q < QFULL) ? (w + NWAVES * q) : (NT - 1);
        if (q < QFULL || w == 0) {
            const _Float16* wrow = &sW[(nt * 16 + m) * KPAD + hiK * 8];
            ((v8h*)&bf0[q])[0] = *(const v8h*)(wrow);
            ((v8h*)&bf0[q])[1] = *(const v8h*)(wrow + 16);
            ((v8h*)&bf1[q])[0] = *(const v8h*)(wrow + 32);
            ((v8h*)&bf1[q])[1] = *(const v8h*)(wrow + 48);
        }
        const int n = nt * 16 + m;                    // global gate column
        sofs[q] = (n % HID) * 4 + (n / HID) + 8 * hiK * GPITCH;
    }

    // ---- hoist activation gather/scatter addressing ----
    int gofs[7], hofs[7];
    #pragma unroll
    for (int p = 0; p < 7; ++p) {
        int e = tid + NTHREADS * p;
        int ec = (e < ROWS * HID) ? e : 0;            // clamp p=6 stragglers
        int row = ec / HID, j = ec - row * HID;
        gofs[p] = row * GPITCH + j * 4;
        hofs[p] = row * KPAD + j;
    }

    float cst[7];
    #pragma unroll
    for (int p = 0; p < 7; ++p) cst[p] = 0.f;

    const size_t xRowStride = (size_t)T_STEPS * INPUT;

    for (int t = 0; t < T_STEPS; ++t) {
        // stage x_t into A columns [HID..HID+2]
        if (tid < ROWS * INPUT) {
            int r = tid / INPUT, i = tid - r * INPUT;
            float xv = x[(size_t)(rowBase + r) * xRowStride + (size_t)t * INPUT + i];
            sH[r * KPAD + HID + i] = (_Float16)xv;
        }
        if (t + 1 < T_STEPS && tid < ROWS)
            __builtin_prefetch(&x[(size_t)(rowBase + tid) * xRowStride +
                                  (size_t)(t + 1) * INPUT], 0, 1);
        __syncthreads();   // h(t-1) + x(t) visible to all waves' A loads

        // A fragments (16x32 f16: lane half h holds K groups {8h..+7, 16+8h..+7})
        const _Float16* hrow = &sH[m * KPAD + hiK * 8];
        v16h a0, a1;
        ((v8h*)&a0)[0] = *(const v8h*)(hrow);
        ((v8h*)&a0)[1] = *(const v8h*)(hrow + 16);
        ((v8h*)&a1)[0] = *(const v8h*)(hrow + 32);
        ((v8h*)&a1)[1] = *(const v8h*)(hrow + 48);

        // gates = A(16x64) x B(64x208): 3 unconditional tile pairs per wave
        #pragma unroll
        for (int q = 0; q < QFULL; ++q) {
            v8f c = {};
            c = __builtin_amdgcn_wmma_f32_16x16x32_f16(false, a0, false, bf0[q],
                                                       (short)0, c, false, false);
            c = __builtin_amdgcn_wmma_f32_16x16x32_f16(false, a1, false, bf1[q],
                                                       (short)0, c, false, false);
            // C layout: VGPR r -> M = r + 8*laneHalf, N folded into sofs
            #pragma unroll
            for (int r = 0; r < 8; ++r)
                sG[r * GPITCH + sofs[q]] = c[r];
        }
        if (w == 0) {                       // tail tile nt=12 (cols 192..199)
            v8f c = {};
            c = __builtin_amdgcn_wmma_f32_16x16x32_f16(false, a0, false, bf0[QFULL],
                                                       (short)0, c, false, false);
            c = __builtin_amdgcn_wmma_f32_16x16x32_f16(false, a1, false, bf1[QFULL],
                                                       (short)0, c, false, false);
            if (m < 8) {                    // n = 192+m < 200; stores only (EXEC ok)
                #pragma unroll
                for (int r = 0; r < 8; ++r)
                    sG[r * GPITCH + sofs[QFULL]] = c[r];
            }
        }
        __syncthreads();   // gates visible to all lanes

        // activations: i,f,g,o adjacent -> one ds_load_b128 per (row,unit)
        #pragma unroll
        for (int p = 0; p < 6; ++p) {
            v4f gv = *(const v4f*)&sG[gofs[p]];
            float ii = fast_sigmoid(gv.x);
            float ff = fast_sigmoid(gv.y);
            float gt = fast_tanh(gv.z);
            float oo = fast_sigmoid(gv.w);
            float cc = ff * cst[p] + ii * gt;
            cst[p] = cc;
            sH[hofs[p]] = (_Float16)(oo * fast_tanh(cc));
        }
        if (tid < ROWS * HID - 6 * NTHREADS) {        // p = 6 stragglers (tid < 32)
            v4f gv = *(const v4f*)&sG[gofs[6]];
            float ii = fast_sigmoid(gv.x);
            float ff = fast_sigmoid(gv.y);
            float gt = fast_tanh(gv.z);
            float oo = fast_sigmoid(gv.w);
            float cc = ff * cst[6] + ii * gt;
            cst[6] = cc;
            sH[hofs[6]] = (_Float16)(oo * fast_tanh(cc));
        }
        // next iteration's __syncthreads orders these h stores before the A loads
    }
    __syncthreads();

    // ---- final classifier: out[row][cls] = h_last . W_fc[cls] + b_fc[cls] ----
    if (tid < ROWS * NCLS) {
        int row = tid / NCLS, cls = tid - row * NCLS;
        float acc = b_fc[cls];
        #pragma unroll
        for (int j = 0; j < HID; ++j)
            acc += (float)sH[row * KPAD + j] * W_fc[cls * HID + j];
        out[(size_t)(rowBase + row) * NCLS + cls] = acc;
    }
}

extern "C" void kernel_launch(void* const* d_in, const int* in_sizes, int n_in,
                              void* d_out, int out_size, void* d_ws, size_t ws_size,
                              hipStream_t stream) {
    const float* x    = (const float*)d_in[0];
    const float* W_ih = (const float*)d_in[1];
    const float* W_hh = (const float*)d_in[2];
    const float* b_ih = (const float*)d_in[3];
    const float* b_hh = (const float*)d_in[4];
    const float* W_fc = (const float*)d_in[5];
    const float* b_fc = (const float*)d_in[6];
    float* out = (float*)d_out;

    const int B = in_sizes[0] / (T_STEPS * INPUT);   // 4096
    const int grid = B / ROWS;                       // 256 workgroups, 4 waves each
    lstm_wmma_kernel<<<grid, NTHREADS, 0, stream>>>(x, W_ih, W_hh, b_ih, b_hh,
                                                    W_fc, b_fc, out);
}
